// Embedding_2937757631008
// MI455X (gfx1250) — matrix operand order repaired and checked by reading, not compiled, not run on hardware
//
#include <hip/hip_runtime.h>
#include <hip/hip_bf16.h>
#include <math.h>

typedef __attribute__((ext_vector_type(16))) _Float16 v16h;
typedef __attribute__((ext_vector_type(8)))  float    v8f;
typedef __attribute__((ext_vector_type(2)))  float    v2f;

#define WPB   4      // waves per block (128 threads, wave32)
#define TPK   208    // true K of tp GEMM (16*13)
#define KK7   7      // ceil(208/32) K-steps of 32

// ---------------------------------------------------------------------------
// Edge kernel: one wave handles 16 edges.
//  A-frag layout (16-bit A 16x32): lane L holds row m=L&15; slot t -> K =
//  (t<8?t:t+8) + 8*(L>>4).  B swizzled in LDS as [tile][lane][slot] with
//  lane = n + 16*((klocal>>3)&1), slot = (klocal&7) + 8*(klocal>=16) (same K order as A).
// ---------------------------------------------------------------------------
__global__ __launch_bounds__(128) void edge_kernel(
    const float* __restrict__ pos, const float* __restrict__ x,
    const float* __restrict__ eain, const int* __restrict__ srcI,
    const int* __restrict__ dstI, const float* __restrict__ w1,
    const float* __restrict__ w2, const float* __restrict__ wtp,
    float* __restrict__ agg, float* __restrict__ oEA, float* __restrict__ oEMB,
    float* __restrict__ oSRC, float* __restrict__ oDST, int E)
{
  __shared__ __align__(32) _Float16 Btp[KK7 * 4 * 32 * 16]; // w_tp swizzled
  __shared__ __align__(32) _Float16 B1[2 * 32 * 16];        // w1 swizzled (K pad 10->32)
  __shared__ __align__(32) _Float16 B2[4 * 32 * 16];        // w2 swizzled
  __shared__ _Float16 hS[WPB][16 * 32];                     // relu hidden, per wave
  __shared__ int dstS[WPB][16];

  const int tid = threadIdx.x;

  // --- cooperative B swizzle into LDS (weights stay hot in L2 across blocks)
  for (int idx = tid; idx < KK7 * 4 * 512; idx += 128) {
    int slot = idx & 15, lane = (idx >> 4) & 31, nt = (idx >> 9) & 3, kk = idx >> 11;
    int k = kk * 32 + ((slot < 8) ? slot : (slot + 8)) + ((lane >> 4) << 3);
    int n = nt * 16 + (lane & 15);
    Btp[idx] = (k < TPK) ? (_Float16)wtp[k * 64 + n] : (_Float16)0.f;
  }
  for (int idx = tid; idx < 2 * 512; idx += 128) {
    int slot = idx & 15, lane = (idx >> 4) & 31, nt = idx >> 9;
    int k = ((slot < 8) ? slot : (slot + 8)) + ((lane >> 4) << 3);
    int n = nt * 16 + (lane & 15);
    B1[idx] = (k < 10) ? (_Float16)w1[k * 32 + n] : (_Float16)0.f;
  }
  for (int idx = tid; idx < 4 * 512; idx += 128) {
    int slot = idx & 15, lane = (idx >> 4) & 31, nt = idx >> 9;
    int k = ((slot < 8) ? slot : (slot + 8)) + ((lane >> 4) << 3);
    int n = nt * 16 + (lane & 15);
    B2[idx] = (_Float16)w2[k * 64 + n];
  }
  __syncthreads();

  const int wave = tid >> 5, lane = tid & 31;
  const int m = lane & 15, khalf = lane >> 4;
  const long ebase = (long)(blockIdx.x * WPB + wave) * 16;
  const long e = ebase + m;
  const bool valid = e < (long)E;
  const long es = valid ? e : 0;

  // --- per-edge scalar math (lane pairs L and L+16 duplicate edge m=L&15)
  const int s = srcI[es], d = dstI[es];
  const float vx = pos[(long)s * 3 + 0] - pos[(long)d * 3 + 0];
  const float vy = pos[(long)s * 3 + 1] - pos[(long)d * 3 + 1];
  const float vz = pos[(long)s * 3 + 2] - pos[(long)d * 3 + 2];
  const float r  = sqrtf(vx * vx + vy * vy + vz * vz);
  const float iv = 1.0f / (r + 1e-12f);
  const float ux = vx * iv, uy = vy * iv, uz = vz * iv;
  const float s3 = 1.7320508075688772f, s5 = 2.23606797749979f, s15 = 3.872983346207417f;
  float ea[13];
  ea[0] = eain[es * 4 + 0]; ea[1] = eain[es * 4 + 1];
  ea[2] = eain[es * 4 + 2]; ea[3] = eain[es * 4 + 3];
  ea[4] = 1.0f; ea[5] = s3 * ux; ea[6] = s3 * uy; ea[7] = s3 * uz;
  ea[8] = s15 * ux * uy; ea[9] = s15 * uy * uz;
  ea[10] = 0.5f * s5 * (3.f * uz * uz - 1.f);
  ea[11] = s15 * ux * uz; ea[12] = 0.5f * s15 * (ux * ux - uy * uy);

  float emb[10];
  const float step = 3.5f / 11.f, sq10 = 3.1622776601683795f;
#pragma unroll
  for (int i = 0; i < 10; ++i) {
    float diff = (r - step * (float)(i + 1)) / step;
    bool inside = (diff > -1.f) && (diff < 1.f);
    emb[i] = inside ? (cosf(1.5707963267948966f * diff) * sq10) : 0.f;
  }
  float xs[16];
#pragma unroll
  for (int i = 0; i < 16; ++i) xs[i] = x[(long)s * 16 + i];

  // --- per-edge outputs (split between lane halves)
  if (valid) {
    if (khalf == 0) {
#pragma unroll
      for (int j = 0; j < 13; ++j) oEA[e * 13 + j] = ea[j];
      dstS[wave][m] = d;
    } else {
#pragma unroll
      for (int i = 0; i < 10; ++i) oEMB[e * 10 + i] = emb[i];
      oSRC[e] = (float)s;
      oDST[e] = (float)d;
    }
  }

  // --- radial MLP, GEMM1: [16x10pad32] @ w1[32x32]
  v16h afe;
  if (khalf == 0) {
#pragma unroll
    for (int t = 0; t < 16; ++t) { int K = (t < 8) ? t : (t + 8);
      afe[t] = (K < 10) ? (_Float16)emb[K] : (_Float16)0.f; }
  } else {
#pragma unroll
    for (int t = 0; t < 16; ++t) { int K = ((t < 8) ? t : (t + 8)) + 8;
      afe[t] = (K < 10) ? (_Float16)emb[K] : (_Float16)0.f; }
  }
  v8f h0 = {}, h1 = {};
  {
    v16h b0 = *(const v16h*)&B1[(0 * 32 + lane) * 16];
    v16h b1 = *(const v16h*)&B1[(1 * 32 + lane) * 16];
    h0 = __builtin_amdgcn_wmma_f32_16x16x32_f16(false, afe, false, b0, (short)0, h0, false, false);
    h1 = __builtin_amdgcn_wmma_f32_16x16x32_f16(false, afe, false, b1, (short)0, h1, false, false);
  }
  // relu -> LDS (C-layout to A-layout bounce, wave-local, LDS is in-order)
#pragma unroll
  for (int r2 = 0; r2 < 8; ++r2) {
    int row = r2 + 8 * khalf;
    hS[wave][row * 32 + m]      = (_Float16)fmaxf(h0[r2], 0.f);
    hS[wave][row * 32 + m + 16] = (_Float16)fmaxf(h1[r2], 0.f);
  }
  v16h afh;
#pragma unroll
  for (int t = 0; t < 16; ++t) {
    int K = ((t < 8) ? t : (t + 8)) + 8 * khalf;
    afh[t] = hS[wave][m * 32 + K];
  }
  // GEMM2: [16x32] @ w2[32x64] -> w-tile, same C-layout as msg accumulators
  v8f wacc[4];
#pragma unroll
  for (int nt = 0; nt < 4; ++nt) {
    v16h b = *(const v16h*)&B2[(nt * 32 + lane) * 16];
    v8f z = {};
    wacc[nt] = __builtin_amdgcn_wmma_f32_16x16x32_f16(false, afh, false, b, (short)0, z, false, false);
  }

  // --- tensor-product GEMM: [16x208pad224] @ w_tp[224x64]
  v8f macc[4] = {{}, {}, {}, {}};
#pragma unroll
  for (int kk = 0; kk < KK7; ++kk) {
    v16h a;
    if (khalf == 0) {
#pragma unroll
      for (int t = 0; t < 16; ++t) {
        int K = kk * 32 + ((t < 8) ? t : (t + 8));
        a[t] = (K < TPK) ? (_Float16)(xs[K / 13] * ea[K % 13]) : (_Float16)0.f;
      }
    } else {
#pragma unroll
      for (int t = 0; t < 16; ++t) {
        int K = kk * 32 + ((t < 8) ? t : (t + 8)) + 8;
        a[t] = (K < TPK) ? (_Float16)(xs[K / 13] * ea[K % 13]) : (_Float16)0.f;
      }
    }
#pragma unroll
    for (int nt = 0; nt < 4; ++nt) {
      v16h b = *(const v16h*)&Btp[((kk * 4 + nt) * 32 + lane) * 16];
      macc[nt] = __builtin_amdgcn_wmma_f32_16x16x32_f16(false, a, false, b, (short)0, macc[nt], false, false);
    }
  }

  // --- msg = macc * wacc, scatter-add into agg[dst][64]
#pragma unroll
  for (int nt = 0; nt < 4; ++nt) {
#pragma unroll
    for (int r2 = 0; r2 < 8; ++r2) {
      int row = r2 + 8 * khalf;
      long erow = ebase + row;
      if (erow < (long)E) {
        float v = macc[nt][r2] * wacc[nt][r2];
        float* p = &agg[(long)dstS[wave][row] * 64 + m + 16 * nt];
        __hip_atomic_fetch_add(p, v, __ATOMIC_RELAXED, __HIP_MEMORY_SCOPE_AGENT);
      }
    }
  }
}

// ---------------------------------------------------------------------------
// Node kernel: node_out = (agg*0.25) @ w_out + einsum(x,na,w_sc), via
// full-precision V_WMMA_F32_16X16X4_F32, both GEMMs into the same accumulators.
// ---------------------------------------------------------------------------
__global__ __launch_bounds__(128) void node_kernel(
    const float* __restrict__ x, const float* __restrict__ nattr,
    const int* __restrict__ batch, const float* __restrict__ wsc,
    const float* __restrict__ wout, const float* __restrict__ agg,
    float* __restrict__ oNode, float* __restrict__ oNA,
    float* __restrict__ oBatch, int N)
{
  __shared__ float Wout[64 * 32];
  __shared__ float Wsc[128 * 32];
  const int tid = threadIdx.x;
  for (int i = tid; i < 64 * 32; i += 128)  Wout[i] = wout[i];
  for (int i = tid; i < 128 * 32; i += 128) Wsc[i]  = wsc[i]; // w_sc already [i*8+j][32]
  __syncthreads();

  const int wave = tid >> 5, lane = tid & 31;
  const int m = lane & 15, khalf = lane >> 4;
  const long nbase = (long)(blockIdx.x * WPB + wave) * 16;
  const long node = nbase + m;
  const bool valid = node < (long)N;
  const long ns = valid ? node : 0;

  float xv[16], nav[8];
#pragma unroll
  for (int i = 0; i < 16; ++i) xv[i] = x[ns * 16 + i];
#pragma unroll
  for (int j = 0; j < 8; ++j)  nav[j] = nattr[ns * 8 + j];

  if (valid && khalf == 0) {
#pragma unroll
    for (int j = 0; j < 8; ++j) oNA[node * 8 + j] = nav[j];
    oBatch[node] = (float)batch[ns];
  }

  v8f acc0 = {}, acc1 = {};

  // (agg/sqrt(16)) @ w_out : K=64 in 16 steps of 4 (f32 WMMA)
#pragma unroll
  for (int ks = 0; ks < 16; ++ks) {
    int kb = ks * 4 + 2 * khalf;
    v2f a; a.x = 0.25f * agg[ns * 64 + kb]; a.y = 0.25f * agg[ns * 64 + kb + 1];
    v2f b0, b1;
    b0.x = Wout[kb * 32 + m];        b0.y = Wout[(kb + 1) * 32 + m];
    b1.x = Wout[kb * 32 + m + 16];   b1.y = Wout[(kb + 1) * 32 + m + 16];
    acc0 = __builtin_amdgcn_wmma_f32_16x16x4_f32(false, a, false, b0, (short)0, acc0, false, false);
    acc1 = __builtin_amdgcn_wmma_f32_16x16x4_f32(false, a, false, b1, (short)0, acc1, false, false);
  }

  // sc = outer(x,na) @ w_sc : K=128 in 32 steps of 4 (f32 WMMA), same acc
#pragma unroll
  for (int ks = 0; ks < 32; ++ks) {
    v2f a;
    if (khalf == 0) {
      int K0 = ks * 4, K1 = ks * 4 + 1;
      a.x = xv[K0 >> 3] * nav[K0 & 7]; a.y = xv[K1 >> 3] * nav[K1 & 7];
    } else {
      int K0 = ks * 4 + 2, K1 = ks * 4 + 3;
      a.x = xv[K0 >> 3] * nav[K0 & 7]; a.y = xv[K1 >> 3] * nav[K1 & 7];
    }
    int kb = ks * 4 + 2 * khalf;
    v2f b0, b1;
    b0.x = Wsc[kb * 32 + m];        b0.y = Wsc[(kb + 1) * 32 + m];
    b1.x = Wsc[kb * 32 + m + 16];   b1.y = Wsc[(kb + 1) * 32 + m + 16];
    acc0 = __builtin_amdgcn_wmma_f32_16x16x4_f32(false, a, false, b0, (short)0, acc0, false, false);
    acc1 = __builtin_amdgcn_wmma_f32_16x16x4_f32(false, a, false, b1, (short)0, acc1, false, false);
  }

#pragma unroll
  for (int r2 = 0; r2 < 8; ++r2) {
    long nrow = nbase + r2 + 8 * khalf;
    if (nrow < (long)N) {
      oNode[nrow * 32 + m]      = acc0[r2];
      oNode[nrow * 32 + m + 16] = acc1[r2];
    }
  }
}

// ---------------------------------------------------------------------------
extern "C" void kernel_launch(void* const* d_in, const int* in_sizes, int n_in,
                              void* d_out, int out_size, void* d_ws, size_t ws_size,
                              hipStream_t stream)
{
  const float* pos   = (const float*)d_in[0];
  const float* x     = (const float*)d_in[1];
  const float* nattr = (const float*)d_in[2];
  const float* eain  = (const float*)d_in[3];
  const int*   eidx  = (const int*)d_in[4];
  const int*   batch = (const int*)d_in[5];
  const float* w1    = (const float*)d_in[6];
  const float* w2    = (const float*)d_in[7];
  const float* wtp   = (const float*)d_in[8];
  const float* wsc   = (const float*)d_in[9];
  const float* wout  = (const float*)d_in[10];

  const int N = in_sizes[0] / 3;
  const int E = in_sizes[3] / 4;
  const int* srcI = eidx;
  const int* dstI = eidx + E;

  float* agg = (float*)d_ws;                         // N*64 floats scratch
  float* out = (float*)d_out;                        // concatenated outputs
  float* oNode  = out;                               // [N,32]
  float* oNA    = oNode + (size_t)N * 32;            // [N,8]
  float* oSRC   = oNA   + (size_t)N * 8;             // [E]
  float* oDST   = oSRC  + (size_t)E;                 // [E]
  float* oEA    = oDST  + (size_t)E;                 // [E,13]
  float* oEMB   = oEA   + (size_t)E * 13;            // [E,10]
  float* oBatch = oEMB  + (size_t)E * 10;            // [N]

  hipMemsetAsync(agg, 0, (size_t)N * 64 * sizeof(float), stream);

  int eblocks = (E + WPB * 16 - 1) / (WPB * 16);
  edge_kernel<<<eblocks, 128, 0, stream>>>(pos, x, eain, srcI, dstI, w1, w2, wtp,
                                           agg, oEA, oEMB, oSRC, oDST, E);

  int nblocks = (N + WPB * 16 - 1) / (WPB * 16);
  node_kernel<<<nblocks, 128, 0, stream>>>(x, nattr, batch, wsc, wout, agg,
                                           oNode, oNA, oBatch, N);
}
